// GATEncoder_10849087389807
// MI455X (gfx1250) — compile-verified, matrix operand order checked by reading, not run on hardware
//
#include <hip/hip_runtime.h>
#include <math.h>

#define NN   50000
#define EE   800000
#define HIDN 128

typedef __attribute__((ext_vector_type(16))) __bf16 v16bf;
typedef __attribute__((ext_vector_type(8)))  float  v8f;

// ---------- helpers ----------
static __device__ inline unsigned short f2bf(float f) {
  unsigned u = __float_as_uint(f);
  unsigned r = (u + 0x7FFFu + ((u >> 16) & 1u)) >> 16;  // round-to-nearest-even
  return (unsigned short)r;
}
// monotonic float<->uint encoding so atomicMax(u32) == float max
static __device__ inline unsigned fenc(float f) {
  unsigned u = __float_as_uint(f);
  return (u & 0x80000000u) ? ~u : (u | 0x80000000u);
}
static __device__ inline float fdec(unsigned u) {
  return (u & 0x80000000u) ? __uint_as_float(u & 0x7FFFFFFFu) : __uint_as_float(~u);
}

// ---------- f32 -> bf16 ----------
__global__ void k_f32_to_bf16(const float* __restrict__ in,
                              unsigned short* __restrict__ out, int n) {
  int i = blockIdx.x * 256 + threadIdx.x;
  if (i < n) out[i] = f2bf(in[i]);
}

// ---------- repack 128x128 f32 weight into WMMA B-fragment layout ----------
// packed[((nt*4+kt)*32+lane)*16+e] : lane-half h=lane>>4, col=nt*16+(lane&15),
// K = kt*32 + 8h + e (+8 if e>=8)   (16-bit B layout, CDNA5 ISA)
__global__ void k_repack_w(const float* __restrict__ W,
                           unsigned short* __restrict__ P) {
  int idx = blockIdx.x * 256 + threadIdx.x;
  if (idx >= 8 * 4 * 32 * 16) return;
  int e    = idx & 15;
  int lane = (idx >> 4) & 31;
  int kt   = (idx >> 9) & 3;
  int nt   = idx >> 11;
  int h = lane >> 4, r = lane & 15;
  int K = kt * 32 + 8 * h + e + (e >= 8 ? 8 : 0);
  int n = nt * 16 + r;
  P[idx] = f2bf(W[K * 128 + n]);
}

// ---------- WMMA GEMM: C[M,128] = A_bf16[M,128] x W(packed) (+bias) ----------
// block = 256 (8 waves); wave w owns column tile nt=w; block owns 16 rows.
// A-strip (16x128 bf16 = 4KB) is staged once per block into LDS with the
// CDNA5 async-to-LDS path, then fragments are read with ds_load_b128.
__global__ void k_gemm_wmma(const unsigned short* __restrict__ Abf,
                            const unsigned short* __restrict__ Pw,
                            const float* __restrict__ bias,
                            float* __restrict__ Cout) {
  __shared__ unsigned short shA[16 * 128];  // sole LDS object -> offset 0
  int m0   = blockIdx.x * 16;
  int tid  = threadIdx.x;
  int wave = tid >> 5;   // = nt, 0..7
  int lane = tid & 31;
  int h = lane >> 4, r = lane & 15;

  // ---- async stage: 256 threads x 16B = 4KB A strip into LDS ----
  {
    unsigned long long gaddr =
        (unsigned long long)(uintptr_t)(Abf + (size_t)m0 * 128 + tid * 8);
    // Derive the LDS address FROM shA so the array escapes into the asm:
    // low 32 bits of the flat address == wave-relative LDS byte offset.
    unsigned lds_off = (unsigned)(uintptr_t)(&shA[tid * 8]);
    asm volatile("global_load_async_to_lds_b128 %0, %1, off"
                 :: "v"(lds_off), "v"(gaddr) : "memory");
    asm volatile("s_wait_asynccnt 0x0" ::: "memory");
  }
  __syncthreads();

  v8f acc = {0.f, 0.f, 0.f, 0.f, 0.f, 0.f, 0.f, 0.f};
  union Frag { v16bf v; uint4 q[2]; };
  const unsigned short* arow = &shA[r * 128];
#pragma unroll
  for (int kt = 0; kt < 4; ++kt) {
    int k0 = kt * 32;
    Frag a, b;
    a.q[0] = *(const uint4*)(arow + k0 + 8 * h);        // K = k0+8h .. +7
    a.q[1] = *(const uint4*)(arow + k0 + 16 + 8 * h);   // K = k0+8h+16 .. +7
    const unsigned short* bp = Pw + (((wave * 4 + kt) * 32 + lane) << 4);
    b.q[0] = *(const uint4*)(bp);
    b.q[1] = *(const uint4*)(bp + 8);
    acc = __builtin_amdgcn_wmma_f32_16x16x32_bf16(
        false, a.v, false, b.v, (short)0, acc, false, false);
  }
  int col = wave * 16 + r;
  float bv = bias ? bias[col] : 0.0f;
#pragma unroll
  for (int v = 0; v < 8; ++v) {
    int orow = m0 + v + 8 * h;                // C layout: lanes16-31 hold M=v+8
    Cout[(size_t)orow * 128 + col] = acc[v] + bv;
  }
}

// ---------- pass A: per-edge message, attention logit, segment max ----------
__global__ void k_edge_logits(const float* __restrict__ xl, const float* __restrict__ xr,
                              const int* __restrict__ ei, const float* __restrict__ ea,
                              const float* __restrict__ We, const float* __restrict__ att,
                              float* __restrict__ logit, unsigned* __restrict__ nmax) {
  int e = blockIdx.x * 8 + (threadIdx.x >> 5);
  if (e >= EE) return;
  int lane = threadIdx.x & 31;
  int src = ei[e];
  int dst = ei[EE + e];
  float a0 = ea[e * 3 + 0], a1 = ea[e * 3 + 1], a2 = ea[e * 3 + 2];
  const float* xls = xl + (size_t)src * 128;
  const float* xrd = xr + (size_t)dst * 128;
  if (lane == 0) {  // software prefetch of the gathered rows -> global_prefetch_b8
    __builtin_prefetch(xls, 0, 0);
    __builtin_prefetch(xrd, 0, 0);
  }
#pragma unroll
  for (int hh = 0; hh < 4; ++hh) {
    int col = hh * 32 + lane;
    float ehc = a0 * We[col] + a1 * We[128 + col] + a2 * We[256 + col];
    float m = xls[col] + xrd[col] + ehc;
    m = m > 0.f ? m : 0.2f * m;                 // leaky_relu(0.2)
    float t = m * att[col];
#pragma unroll
    for (int off = 16; off; off >>= 1) t += __shfl_xor(t, off, 32);
    if (lane == 0) {
      logit[(size_t)e * 4 + hh] = t;
      atomicMax(&nmax[(size_t)dst * 4 + hh], fenc(t));
    }
  }
}

// ---------- pass B: p = exp(logit - max), accumulate denominator ----------
__global__ void k_edge_softmax_p(float* __restrict__ logit, const unsigned* __restrict__ nmax,
                                 float* __restrict__ den, const int* __restrict__ ei) {
  int e = blockIdx.x * 256 + threadIdx.x;   // one thread per edge, float4 logits
  if (e >= EE) return;
  int dst = ei[EE + e];
  float4 lg = *(float4*)(logit + (size_t)e * 4);
  const unsigned* nm = nmax + (size_t)dst * 4;
  float4 pv;
  pv.x = __expf(lg.x - fdec(nm[0]));
  pv.y = __expf(lg.y - fdec(nm[1]));
  pv.z = __expf(lg.z - fdec(nm[2]));
  pv.w = __expf(lg.w - fdec(nm[3]));
  *(float4*)(logit + (size_t)e * 4) = pv;   // in place
  float* dd = den + (size_t)dst * 4;
  atomicAdd(dd + 0, pv.x);
  atomicAdd(dd + 1, pv.y);
  atomicAdd(dd + 2, pv.z);
  atomicAdd(dd + 3, pv.w);
}

// ---------- pass C: acc[dst] += alpha * xl[src] ----------
__global__ void k_edge_aggregate(const float* __restrict__ p, const float* __restrict__ den,
                                 const float* __restrict__ xl, const int* __restrict__ ei,
                                 float* __restrict__ acc) {
  int e = blockIdx.x * 8 + (threadIdx.x >> 5);
  if (e >= EE) return;
  int lane = threadIdx.x & 31;
  int src = ei[e], dst = ei[EE + e];
  const float* xls = xl + (size_t)src * 128;
  float* accd = acc + (size_t)dst * 128;
#pragma unroll
  for (int hh = 0; hh < 4; ++hh) {
    float alpha = p[(size_t)e * 4 + hh] / (den[(size_t)dst * 4 + hh] + 1e-16f);
    atomicAdd(&accd[hh * 32 + lane], alpha * xls[hh * 32 + lane]);
  }
}

// ---------- graph-LayerNorm stats: sum / sumsq over all N*HID ----------
__global__ void k_reduce_stats(const float* __restrict__ acc, float* __restrict__ scal, int n) {
  float s1 = 0.f, s2 = 0.f;
  for (int i = blockIdx.x * 256 + threadIdx.x; i < n; i += gridDim.x * 256) {
    float v = acc[i];
    s1 += v;
    s2 += v * v;
  }
  for (int off = 16; off; off >>= 1) {
    s1 += __shfl_xor(s1, off, 32);
    s2 += __shfl_xor(s2, off, 32);
  }
  __shared__ float sh1[8], sh2[8];
  int lane = threadIdx.x & 31, w = threadIdx.x >> 5;
  if (lane == 0) { sh1[w] = s1; sh2[w] = s2; }
  __syncthreads();
  if (threadIdx.x == 0) {
    float t1 = 0.f, t2 = 0.f;
    for (int i = 0; i < 8; ++i) { t1 += sh1[i]; t2 += sh2[i]; }
    atomicAdd(&scal[0], t1);
    atomicAdd(&scal[1], t2);
  }
}

// ---------- apply LN (graph mode) + exact-erf GELU, emit bf16 ----------
__global__ void k_ln_gelu_bf16(const float* __restrict__ acc, const float* __restrict__ scal,
                               const float* __restrict__ lnw, const float* __restrict__ lnb,
                               unsigned short* __restrict__ xbf, int n) {
  int i = blockIdx.x * 256 + threadIdx.x;
  if (i >= n) return;
  float invM = 1.0f / (float)n;
  float mean = scal[0] * invM;
  float ms = fmaxf(scal[1] * invM - mean * mean, 0.0f);  // mean((x-mean)^2)
  float dinv = 1.0f / (sqrtf(ms) + 1e-5f);
  int col = i & 127;
  float y = (acc[i] - mean) * dinv;
  y = y * lnw[col] + lnb[col];
  float g = 0.5f * y * (1.0f + erff(y * 0.70710678118654752f));  // exact gelu
  xbf[i] = f2bf(g);
}

// =====================================================================
extern "C" void kernel_launch(void* const* d_in, const int* in_sizes, int n_in,
                              void* d_out, int out_size, void* d_ws, size_t ws_size,
                              hipStream_t stream) {
  const float* x    = (const float*)d_in[0];
  const int*   ei   = (const int*)d_in[1];
  const float* ea   = (const float*)d_in[2];
  const float* Wl   = (const float*)d_in[3];   // [2,128,128]
  const float* Wr   = (const float*)d_in[4];   // [2,128,128]
  const float* We   = (const float*)d_in[5];   // [2,3,128]
  const float* att  = (const float*)d_in[6];   // [2,4,32]
  const float* lnw  = (const float*)d_in[7];   // [2,128]
  const float* lnb  = (const float*)d_in[8];   // [2,128]
  const float* linw = (const float*)d_in[9];   // [128,128]
  const float* linb = (const float*)d_in[10];  // [128]
  float* out = (float*)d_out;

  // workspace carve-out (256B aligned slices)
  char* wsp = (char*)d_ws;
  size_t o = 0;
  auto carve = [&](size_t bytes) -> void* {
    void* p = wsp + o;
    o += (bytes + 255) & ~(size_t)255;
    return p;
  };
  float*          xl    = (float*)carve((size_t)NN * 128 * 4);
  float*          xracc = (float*)carve((size_t)NN * 128 * 4);  // xr, then acc
  unsigned short* xbf   = (unsigned short*)carve((size_t)NN * 128 * 2);
  float*          logit = (float*)carve((size_t)EE * 4 * 4);    // logits, then p
  unsigned*       nmax  = (unsigned*)carve((size_t)NN * 4 * 4);
  float*          den   = (float*)carve((size_t)NN * 4 * 4);
  unsigned short* pw    = (unsigned short*)carve((size_t)5 * 16384 * 2);
  float*          scal  = (float*)carve(256);

  // one-time: x -> bf16, repack all 5 weight matrices into WMMA B layout
  k_f32_to_bf16<<<(NN * 128 + 255) / 256, 256, 0, stream>>>(x, xbf, NN * 128);
  k_repack_w<<<64, 256, 0, stream>>>(Wl,          pw + 0 * 16384);  // Wl layer0
  k_repack_w<<<64, 256, 0, stream>>>(Wl + 16384,  pw + 1 * 16384);  // Wl layer1
  k_repack_w<<<64, 256, 0, stream>>>(Wr,          pw + 2 * 16384);  // Wr layer0
  k_repack_w<<<64, 256, 0, stream>>>(Wr + 16384,  pw + 3 * 16384);  // Wr layer1
  k_repack_w<<<64, 256, 0, stream>>>(linw,        pw + 4 * 16384);  // final lin

  for (int l = 0; l < 2; ++l) {
    // dense projections (WMMA bf16 -> f32)
    k_gemm_wmma<<<NN / 16, 256, 0, stream>>>(xbf, pw + l * 16384,       nullptr, xl);
    k_gemm_wmma<<<NN / 16, 256, 0, stream>>>(xbf, pw + (2 + l) * 16384, nullptr, xracc);

    hipMemsetAsync(nmax, 0, (size_t)NN * 4 * 4, stream);  // fdec(0) never read (only nodes w/ edges)
    hipMemsetAsync(den,  0, (size_t)NN * 4 * 4, stream);

    k_edge_logits<<<(EE + 7) / 8, 256, 0, stream>>>(
        xl, xracc, ei, ea, We + l * 384, att + l * 128, logit, nmax);
    k_edge_softmax_p<<<(EE + 255) / 256, 256, 0, stream>>>(logit, nmax, den, ei);

    hipMemsetAsync(xracc, 0, (size_t)NN * 128 * 4, stream);  // xr dead, reuse as acc
    k_edge_aggregate<<<(EE + 7) / 8, 256, 0, stream>>>(logit, den, xl, ei, xracc);

    hipMemsetAsync(scal, 0, 8, stream);
    k_reduce_stats<<<1024, 256, 0, stream>>>(xracc, scal, NN * 128);
    k_ln_gelu_bf16<<<(NN * 128 + 255) / 256, 256, 0, stream>>>(
        xracc, scal, lnw + l * 128, lnb + l * 128, xbf, NN * 128);
  }

  // final jk Linear with bias -> d_out (f32)
  k_gemm_wmma<<<NN / 16, 256, 0, stream>>>(xbf, pw + 4 * 16384, linb, out);
}
